// LSTMClassifier_12189117186064
// MI455X (gfx1250) — matrix operand order repaired and checked by reading, not compiled, not run on hardware
//
#include <hip/hip_runtime.h>
#include <stdint.h>

typedef __bf16 bf16;
typedef __attribute__((ext_vector_type(8)))  bf16  v8bf;
typedef __attribute__((ext_vector_type(16))) bf16  v16bf;
typedef __attribute__((ext_vector_type(8)))  float v8f;
typedef __attribute__((ext_vector_type(4)))  unsigned int u32x4;
typedef __attribute__((ext_vector_type(8)))  int i32x8;
typedef __attribute__((ext_vector_type(4)))  int i32x4;

#define B_  128
#define S_  256
#define I_  1024
#define H_  1024
#define G_  4096   // 4*H
#define C_  1000

// ---------------------------------------------------------------- helpers

__device__ __forceinline__ bf16 f2bf(float f) {
  unsigned u = __builtin_bit_cast(unsigned, f);
  u = (u + 0x7FFFu + ((u >> 16) & 1u)) >> 16;
  unsigned short s = (unsigned short)u;
  return __builtin_bit_cast(bf16, s);
}

// CDNA5 async global->LDS copy (16B per lane), tracked by ASYNCcnt.
__device__ __forceinline__ void async_b128(void* lds_ptr, const void* gptr) {
  unsigned lds = (unsigned)(unsigned long long)lds_ptr;
  asm volatile("global_load_async_to_lds_b128 %0, %1, off"
               :: "v"(lds), "v"(gptr) : "memory");
}
__device__ __forceinline__ void wait_async0() {
  asm volatile("s_wait_asynccnt 0x0" ::: "memory");
}

// CDNA5 Tensor Data Mover: DMA a (up to 3D) bf16 tile global->LDS, tracked by
// TENSORcnt. D# built per ISA 8.3-8.6: dims/strides in 2-byte elements; tile
// lands row-major (x innermost) at lds_addr, matching our [z][y][x] LDS tiles.
// This toolchain exposes the 6-arg builtin: (g0, g1, g2, g3, g4, cpol).
__device__ __forceinline__ void tdm_load_bf16(unsigned lds_addr, const void* gptr,
                                              unsigned tile_d0, unsigned tile_d1,
                                              unsigned tile_d2,
                                              unsigned long long stride0_el,
                                              unsigned long long stride1_el,
                                              unsigned td0, unsigned td1, unsigned td2) {
  unsigned long long ga = (unsigned long long)gptr;
  u32x4 g0;
  g0[0] = 1u;                                                   // count=1 (valid user D#)
  g0[1] = lds_addr;                                             // LDS byte address
  g0[2] = (unsigned)ga;                                         // global_addr[31:0]
  g0[3] = (unsigned)((ga >> 32) & 0x01FFFFFFu) | (2u << 30);    // addr[56:32] | type=2
  i32x8 g1;
  g1[0] = (int)(1u << 16);                                      // data_size=1 (2B), no multicast
  g1[1] = (int)((td0 & 0xFFFFu) << 16);                         // tensor_dim0[15:0]
  g1[2] = (int)(((td0 >> 16) & 0xFFFFu) | ((td1 & 0xFFFFu) << 16));
  g1[3] = (int)(((td1 >> 16) & 0xFFFFu) | ((tile_d0 & 0xFFFFu) << 16));
  g1[4] = (int)((tile_d1 & 0xFFFFu) | ((tile_d2 & 0xFFFFu) << 16));
  g1[5] = (int)(stride0_el & 0xFFFFFFFFull);                    // tensor_dim0_stride[31:0]
  g1[6] = (int)(((stride0_el >> 32) & 0xFFFFull) | ((stride1_el & 0xFFFFull) << 16));
  g1[7] = (int)((stride1_el >> 16) & 0xFFFFFFFFull);            // tensor_dim1_stride[47:16]
  i32x4 g2; g2[0] = (int)td2; g2[1] = 0; g2[2] = 0; g2[3] = 0;  // tensor_dim2
  i32x4 g3; g3[0] = 0; g3[1] = 0; g3[2] = 0; g3[3] = 0;
  i32x8 g4;                                                     // unused trailing group
  g4[0] = 0; g4[1] = 0; g4[2] = 0; g4[3] = 0;
  g4[4] = 0; g4[5] = 0; g4[6] = 0; g4[7] = 0;
  __builtin_amdgcn_tensor_load_to_lds(g0, g1, g2, g3, g4, 0);
}
__device__ __forceinline__ void wait_tensor0() {
  __builtin_amdgcn_s_wait_tensorcnt((short)0);
}

// A fragment (16x32 bf16, ISA 7.12.2): lane L<16 holds row M=L, K={0..7,16..23};
// lane L>=16 holds row M=L-16, K={8..15,24..31}.
__device__ __forceinline__ v16bf ldfragA(const bf16* p) {
  v8bf lo = *(const v8bf*)p;
  v8bf hi = *(const v8bf*)(p + 16);
  return __builtin_shufflevector(lo, hi, 0,1,2,3,4,5,6,7,8,9,10,11,12,13,14,15);
}
// B fragment (32x16 bf16): lane L<16 holds col N=L, K={0..7,16..23}; lane L>=16 holds
// N=L-16, K={8..15,24..31}. W tile stored [n][k] row-major.
__device__ __forceinline__ v16bf ldfragB(const bf16* p) {
  v8bf lo = *(const v8bf*)p;
  v8bf hi = *(const v8bf*)(p + 16);
  return __builtin_shufflevector(lo, hi, 0,1,2,3,4,5,6,7,8,9,10,11,12,13,14,15);
}

#define WMMA_BF16(a, b, c) \
  __builtin_amdgcn_wmma_f32_16x16x32_bf16(false, (a), false, (b), (short)0, (c), false, false)

__device__ __forceinline__ float sigmoidf_(float x) { return 1.0f / (1.0f + __expf(-x)); }

// ---------------------------------------------------------------- prep kernels

__global__ __launch_bounds__(256) void cvt4_f32_bf16(const float* __restrict__ src,
                                                     bf16* __restrict__ dst, int n4) {
  int i = blockIdx.x * 256 + threadIdx.x;
  if (i < n4) {
    float4 v = ((const float4*)src)[i];
    ushort4 pk;
    pk.x = __builtin_bit_cast(unsigned short, f2bf(v.x));
    pk.y = __builtin_bit_cast(unsigned short, f2bf(v.y));
    pk.z = __builtin_bit_cast(unsigned short, f2bf(v.z));
    pk.w = __builtin_bit_cast(unsigned short, f2bf(v.w));
    ((ushort4*)dst)[i] = pk;
  }
}

// Pad Wfc [1000][1024] -> bf16 [1024][1024] (rows >= 1000 zero).
__global__ __launch_bounds__(256) void pad_wfc(const float* __restrict__ Wfc,
                                               bf16* __restrict__ dst) {
  int i   = blockIdx.x * 256 + threadIdx.x;   // 0 .. 1024*1024-1
  int row = i >> 10, col = i & 1023;
  dst[i] = (row < C_) ? f2bf(Wfc[row * 1024 + col]) : f2bf(0.0f);
}

// bias = bxh + bhh ; h0 = 0 (bf16) ; c = 0 (f32)
__global__ __launch_bounds__(256) void prep_state(const float* __restrict__ bxh,
                                                  const float* __restrict__ bhh,
                                                  float* __restrict__ bias,
                                                  bf16* __restrict__ h0,
                                                  float* __restrict__ c0) {
  int i = blockIdx.x * 256 + threadIdx.x;     // 0 .. B_*H_-1
  if (i < G_) bias[i] = bxh[i] + bhh[i];
  if (i < B_ * H_) { h0[i] = f2bf(0.0f); c0[i] = 0.0f; }
}

// ---------------------------------------------------------------- phase 1: xW GEMM
// xW[m, g] = sum_k xb[m,k] * Wxh[g,k] + bias[g],  m = s*128 + b  (xW is [S][B][4H]).
// A tile (x rows, fixed s per 128-row block) via per-lane async-to-LDS;
// B tile (128x32 weight block) via one TDM descriptor issued by wave 0.

__global__ __launch_bounds__(256) void gemm_xw(const bf16* __restrict__ Xb,   // [(b*S+s)][I]
                                               const bf16* __restrict__ Wb,   // [4H][I]
                                               const float* __restrict__ bias,
                                               float* __restrict__ xW) {
  __shared__ bf16 As[2][128][32];
  __shared__ bf16 Bs[2][128][32];
  const int tid  = threadIdx.x;
  const int lane = tid & 31;
  const int wave = tid >> 5;
  const int m0 = blockIdx.x * 128;
  const int n0 = blockIdx.y * 128;
  const int wm = wave & 1;
  const int wn = wave >> 1;
  const int r  = lane & 15;
  const int hf = lane >> 4;
  const int s_idx = m0 >> 7;      // fixed timestep for this 128-row block

  auto issueA = [&](int buf, int k0) {
#pragma unroll
    for (int c = tid; c < 512; c += 256) {        // 128x32, 16B chunks
      int row = c >> 2, part = c & 3;
      // m = m0 + row -> b = row, s = s_idx
      async_b128(&As[buf][row][part * 8],
                 Xb + ((size_t)row * S_ + s_idx) * I_ + k0 + part * 8);
    }
  };
  auto issueB = [&](int buf, int k0) {
    if (wave == 0)
      tdm_load_bf16((unsigned)(unsigned long long)(void*)&Bs[buf][0][0],
                    Wb + (size_t)n0 * I_ + k0,
                    /*tile*/ 32, 128, 0, /*strides*/ I_, 0, /*dims*/ I_, G_, 1);
  };

  v8f acc[4][2];
#pragma unroll
  for (int i = 0; i < 4; ++i)
#pragma unroll
    for (int j = 0; j < 2; ++j)
#pragma unroll
      for (int e = 0; e < 8; ++e) acc[i][j][e] = 0.0f;

  issueA(0, 0);
  issueB(0, 0);
  for (int ks = 0; ks < I_ / 32; ++ks) {
    const int cur = ks & 1;
    wait_async0();                       // this wave's A-chunk DMA done
    if (wave == 0) wait_tensor0();       // B-tile TDM done
    __syncthreads();                     // all DMA visible; prior readers drained
    if (ks + 1 < I_ / 32) { issueA(cur ^ 1, (ks + 1) * 32); issueB(cur ^ 1, (ks + 1) * 32); }

    v16bf af[4], bf_[2];
#pragma unroll
    for (int i = 0; i < 4; ++i) af[i]  = ldfragA(&As[cur][wm * 64 + i * 16 + r][hf * 8]);
#pragma unroll
    for (int j = 0; j < 2; ++j) bf_[j] = ldfragB(&Bs[cur][wn * 32 + j * 16 + r][hf * 8]);
#pragma unroll
    for (int i = 0; i < 4; ++i)
#pragma unroll
      for (int j = 0; j < 2; ++j) acc[i][j] = WMMA_BF16(af[i], bf_[j], acc[i][j]);
  }

  // C/D layout: VGPR e -> M = e + 8*hf, N = r
#pragma unroll
  for (int i = 0; i < 4; ++i)
#pragma unroll
    for (int j = 0; j < 2; ++j) {
      int mb = m0 + wm * 64 + i * 16 + hf * 8;
      int nb = n0 + wn * 32 + j * 16 + r;
      float bv = bias[nb];
#pragma unroll
      for (int e = 0; e < 8; ++e)
        xW[(size_t)(mb + e) * G_ + nb] = acc[i][j][e] + bv;
    }
}

// ---------------------------------------------------------------- phase 2: recurrent step
// gates[m, g*H + n] = xW_t[m, g*H+n] + sum_k h_prev[m,k] * Whh[g*H+n, k]
// All 4 gate weight tiles fetched by ONE 3D TDM op (x=32 k, y=32 rows, z=4 gates,
// z-stride = H*H elements) landing exactly in Bs[buf][4][32][32].

__global__ __launch_bounds__(256) void lstm_step(const bf16* __restrict__ Hprev, // [128][1024]
                                                 bf16* __restrict__ Hnext,       // [128][1024]
                                                 float* __restrict__ Cst,        // [128][1024]
                                                 const bf16* __restrict__ Whh,   // [4096][1024]
                                                 const float* __restrict__ xWt)  // [128][4096]
{
  __shared__ bf16 As[2][64][32];
  __shared__ bf16 Bs[2][4][32][32];
  const int tid  = threadIdx.x;
  const int lane = tid & 31;
  const int wave = tid >> 5;
  const int m0 = blockIdx.x * 64;
  const int n0 = blockIdx.y * 32;
  const int wm = wave & 3;
  const int wn = wave >> 2;
  const int r  = lane & 15;
  const int hf = lane >> 4;
  const int base_m = m0 + wm * 16;
  const int base_n = n0 + wn * 16;

  auto issueA = [&](int buf, int k0) {
    int row = tid >> 2, part = tid & 3;        // 64x32 = 256 chunks, one per thread
    async_b128(&As[buf][row][part * 8], Hprev + (size_t)(m0 + row) * H_ + k0 + part * 8);
  };
  auto issueB = [&](int buf, int k0) {
    if (wave == 0)
      tdm_load_bf16((unsigned)(unsigned long long)(void*)&Bs[buf][0][0][0],
                    Whh + (size_t)n0 * H_ + k0,
                    /*tile*/ 32, 32, 4,
                    /*strides*/ H_, (unsigned long long)H_ * H_,
                    /*dims*/ H_, G_, 4);
  };

  // init accumulators from precomputed input projection (C of D = A*B + C)
  v8f acc[4];
#pragma unroll
  for (int g = 0; g < 4; ++g)
#pragma unroll
    for (int e = 0; e < 8; ++e)
      acc[g][e] = xWt[(size_t)(base_m + hf * 8 + e) * G_ + g * H_ + base_n + r];

  issueA(0, 0);
  issueB(0, 0);
  for (int ks = 0; ks < H_ / 32; ++ks) {
    const int cur = ks & 1;
    wait_async0();
    if (wave == 0) wait_tensor0();
    __syncthreads();
    if (ks + 1 < H_ / 32) { issueA(cur ^ 1, (ks + 1) * 32); issueB(cur ^ 1, (ks + 1) * 32); }

    v16bf af = ldfragA(&As[cur][wm * 16 + r][hf * 8]);
#pragma unroll
    for (int g = 0; g < 4; ++g) {
      v16bf bf_ = ldfragB(&Bs[cur][g][wn * 16 + r][hf * 8]);
      acc[g] = WMMA_BF16(af, bf_, acc[g]);
    }
  }

  // fused LSTM cell update (gate order: i, f, chat, o)
#pragma unroll
  for (int e = 0; e < 8; ++e) {
    int m = base_m + hf * 8 + e;
    int n = base_n + r;
    size_t idx = (size_t)m * H_ + n;
    float it = sigmoidf_(acc[0][e]);
    float ft = sigmoidf_(acc[1][e]);
    float ch = tanhf(acc[2][e]);
    float ot = sigmoidf_(acc[3][e]);
    float cn = Cst[idx] * ft + it * ch;
    Cst[idx] = cn;
    Hnext[idx] = f2bf(ot * tanhf(cn));
  }
}

// ---------------------------------------------------------------- phase 3: FC head

__global__ __launch_bounds__(256) void fc_head(const bf16* __restrict__ Hl,   // [128][1024]
                                               const bf16* __restrict__ Wf,   // [1024][1024]
                                               const float* __restrict__ bfc, // [1000]
                                               float* __restrict__ out)       // [128][1000]
{
  __shared__ bf16 As[2][64][32];
  __shared__ bf16 Bs[2][64][32];
  const int tid  = threadIdx.x;
  const int lane = tid & 31;
  const int wave = tid >> 5;
  const int m0 = blockIdx.x * 64;
  const int n0 = blockIdx.y * 64;
  const int wm = wave & 3;
  const int wv = wave >> 2;
  const int r  = lane & 15;
  const int hf = lane >> 4;

  auto issueA = [&](int buf, int k0) {
    int row = tid >> 2, part = tid & 3;
    async_b128(&As[buf][row][part * 8], Hl + (size_t)(m0 + row) * H_ + k0 + part * 8);
  };
  auto issueB = [&](int buf, int k0) {
    if (wave == 0)
      tdm_load_bf16((unsigned)(unsigned long long)(void*)&Bs[buf][0][0],
                    Wf + (size_t)n0 * H_ + k0,
                    /*tile*/ 32, 64, 0, /*strides*/ H_, 0, /*dims*/ H_, 1024, 1);
  };

  v8f acc[2];
#pragma unroll
  for (int j = 0; j < 2; ++j)
#pragma unroll
    for (int e = 0; e < 8; ++e) acc[j][e] = 0.0f;

  issueA(0, 0);
  issueB(0, 0);
  for (int ks = 0; ks < H_ / 32; ++ks) {
    const int cur = ks & 1;
    wait_async0();
    if (wave == 0) wait_tensor0();
    __syncthreads();
    if (ks + 1 < H_ / 32) { issueA(cur ^ 1, (ks + 1) * 32); issueB(cur ^ 1, (ks + 1) * 32); }

    v16bf af = ldfragA(&As[cur][wm * 16 + r][hf * 8]);
#pragma unroll
    for (int j = 0; j < 2; ++j) {
      v16bf bf_ = ldfragB(&Bs[cur][wv * 32 + j * 16 + r][hf * 8]);
      acc[j] = WMMA_BF16(af, bf_, acc[j]);
    }
  }

#pragma unroll
  for (int j = 0; j < 2; ++j) {
    int mb = m0 + wm * 16 + hf * 8;
    int nb = n0 + wv * 32 + j * 16 + r;
    if (nb < C_) {
      float bv = bfc[nb];
#pragma unroll
      for (int e = 0; e < 8; ++e)
        out[(size_t)(mb + e) * C_ + nb] = acc[j][e] + bv;
    }
  }
}

// ---------------------------------------------------------------- launch

extern "C" void kernel_launch(void* const* d_in, const int* in_sizes, int n_in,
                              void* d_out, int out_size, void* d_ws, size_t ws_size,
                              hipStream_t stream) {
  const float* x   = (const float*)d_in[0];
  const float* Wxh = (const float*)d_in[1];
  const float* bxh = (const float*)d_in[2];
  const float* Whh = (const float*)d_in[3];
  const float* bhh = (const float*)d_in[4];
  const float* Wfc = (const float*)d_in[5];
  const float* bfc = (const float*)d_in[6];
  float* out = (float*)d_out;

  char* ws = (char*)d_ws;
  size_t off = 0;
  bf16*  xb    = (bf16*)(ws + off);  off += (size_t)B_ * S_ * I_ * 2;   // 64 MB
  bf16*  wxhb  = (bf16*)(ws + off);  off += (size_t)G_ * I_ * 2;        // 8 MB
  bf16*  whhb  = (bf16*)(ws + off);  off += (size_t)G_ * H_ * 2;        // 8 MB
  bf16*  wfcb  = (bf16*)(ws + off);  off += (size_t)1024 * 1024 * 2;    // 2 MB
  float* bias  = (float*)(ws + off); off += (size_t)G_ * 4;             // 16 KB
  float* xW    = (float*)(ws + off); off += (size_t)S_ * B_ * G_ * 4;   // 512 MB
  bf16*  h0    = (bf16*)(ws + off);  off += (size_t)B_ * H_ * 2;
  bf16*  h1    = (bf16*)(ws + off);  off += (size_t)B_ * H_ * 2;
  float* cbuf  = (float*)(ws + off); off += (size_t)B_ * H_ * 4;

  // precision conversion + state init (re-done every call: deterministic)
  cvt4_f32_bf16<<<(B_ * S_ * I_ / 4 + 255) / 256, 256, 0, stream>>>(x, xb, B_ * S_ * I_ / 4);
  cvt4_f32_bf16<<<(G_ * I_ / 4 + 255) / 256, 256, 0, stream>>>(Wxh, wxhb, G_ * I_ / 4);
  cvt4_f32_bf16<<<(G_ * H_ / 4 + 255) / 256, 256, 0, stream>>>(Whh, whhb, G_ * H_ / 4);
  pad_wfc<<<(1024 * 1024) / 256, 256, 0, stream>>>(Wfc, wfcb);
  prep_state<<<(B_ * H_) / 256, 256, 0, stream>>>(bxh, bhh, bias, h0, cbuf);

  // phase 1: one big WMMA GEMM for the input projection of all timesteps
  gemm_xw<<<dim3(S_ * B_ / 128, G_ / 128), 256, 0, stream>>>(xb, wxhb, bias, xW);

  // phase 2: 256 fused recurrent steps (double-buffered h)
  bf16* hbuf[2] = {h0, h1};
  for (int t = 0; t < S_; ++t) {
    lstm_step<<<dim3(B_ / 64, H_ / 32), 256, 0, stream>>>(
        hbuf[t & 1], hbuf[(t + 1) & 1], cbuf, whhb, xW + (size_t)t * B_ * G_);
  }

  // phase 3: classifier head on h_last (S_ even -> lives in hbuf[0])
  fc_head<<<dim3(B_ / 64, 1024 / 64), 256, 0, stream>>>(hbuf[S_ & 1], wfcb, bfc, out);
}
